// KernelInterpolationFeat_3934190044192
// MI455X (gfx1250) — matrix operand order repaired and checked by reading, not compiled, not run on hardware
//
#include <hip/hip_runtime.h>
#include <hip/hip_bf16.h>

#define NG (48*48*48)   // 110592 voxels
#define CP 128          // channels
#define NK 512          // keypoints
#define EPSN 1e-5f

typedef __attribute__((ext_vector_type(16))) __bf16 v16bf;
typedef __attribute__((ext_vector_type(8)))  float  v8f;

// K-index mapping inside a 16-bit A-fragment (16x32), matching the B loads:
// lanes 0-15:  element i -> K = 0..15;  lanes 16-31: K = 16..31
__device__ __forceinline__ int a_klocal(int i, int hi) {
    int k = i;
    return hi ? k + 16 : k;
}

// ---------------- weight packing: W0 (co,ci,3,3,3) -> A-fragments ----------------
// frag id = (tap*4 + ks)*8 + mt ; element = lane*16 + i
__global__ void k_pack_w(const float* __restrict__ W0, __bf16* __restrict__ Wp) {
    int idx = blockIdx.x * blockDim.x + threadIdx.x;
    if (idx >= 27 * 4 * 8 * 512) return;
    int i    = idx & 15;
    int lane = (idx >> 4) & 31;
    int frag = idx >> 9;
    int mt   = frag & 7;
    int ks   = (frag >> 3) & 3;
    int tap  = frag >> 5;
    int hi   = lane >> 4;
    int co   = mt * 16 + (lane & 15);
    int ci   = ks * 32 + a_klocal(i, hi);
    Wp[idx] = (__bf16)W0[(co * CP + ci) * 27 + tap];
}

// ---------------- build bf16 input: concat(feats, grid) in [vox][ch] ----------------
__global__ void k_build_in(const float* __restrict__ feats, __bf16* __restrict__ xin) {
    int idx = blockIdx.x * blockDim.x + threadIdx.x;
    if (idx >= NG * CP) return;
    int c = idx & 127;
    int vox = idx >> 7;
    float v;
    if (c < 125) {
        v = feats[c * NG + vox];
    } else {
        int x = vox % 48, y = (vox / 48) % 48, z = vox / 2304;
        int p = (c == 125) ? x : (c == 126) ? y : z;
        v = -1.0f + (2.0f / 47.0f) * (float)p;
    }
    xin[idx] = (__bf16)v;
}

// ---------------- implicit-GEMM conv3d (bf16 WMMA, f32 acc) + ReLU ----------------
// wave: 16 voxels (N) x 128 out-channels (8 M-tiles); K = 27 taps x 128 ci.
// Weights for the current tap staged in LDS (shared by all 8 waves), double-
// buffered, filled with async global->LDS copies overlapping the WMMAs; the
// 32 A-fragment ds_loads per tap are software-pipelined one step ahead of the
// WMMA consuming them.
#define TAP_ELEMS (4 * 8 * 512)          // 16384 bf16 = 32 KB per tap
__global__ void __launch_bounds__(256) k_conv(const __bf16* __restrict__ xin,
                                              const __bf16* __restrict__ Wp,
                                              const float* __restrict__ b0,
                                              float* __restrict__ y) {
    __shared__ __bf16 smem[2 * TAP_ELEMS];   // 64 KB double buffer

    int tid  = threadIdx.x;
    int lane = tid & 31;
    int wave = tid >> 5;
    int vtile = (blockIdx.x * 8 + wave) * 16;
    int hi = lane >> 4;
    int l  = lane & 15;
    int vox = vtile + l;
    int xb = vox % 48;
    int yb = (vox / 48) % 48;   // wave-uniform (16 | 48)
    int zb = vox / 2304;        // wave-uniform

    // async copy of one 32 KB tap block: 256 threads x 128 B each
    auto stage_tap = [&](int tap, int buf) {
        unsigned ldsbase = (unsigned)(uintptr_t)(&smem[buf * TAP_ELEMS]);
        const char* gsrc = (const char*)(Wp + (size_t)tap * TAP_ELEMS);
#pragma unroll
        for (int j = 0; j < 8; ++j) {
            unsigned boff = (unsigned)(tid * 16 + j * 4096);   // bytes
            unsigned lds  = ldsbase + boff;
            unsigned long long ga = (unsigned long long)(uintptr_t)(gsrc + boff);
            asm volatile("global_load_async_to_lds_b128 %0, %1, off"
                         :: "v"(lds), "v"(ga) : "memory");
        }
    };

    v8f acc[8];
#pragma unroll
    for (int m = 0; m < 8; ++m)
#pragma unroll
        for (int r = 0; r < 8; ++r) acc[m][r] = 0.0f;

    // prologue: stage tap 0
    stage_tap(0, 0);
    asm volatile("s_wait_asynccnt 0x0" ::: "memory");
    __syncthreads();

    for (int tap = 0; tap < 27; ++tap) {
        int buf = tap & 1;
        if (tap + 1 < 27) stage_tap(tap + 1, (tap + 1) & 1);   // overlaps WMMAs below

        int dz = tap / 9 - 1, dy = (tap / 3) % 3 - 1, dx = tap % 3 - 1;
        int z2 = zb + dz, y2 = yb + dy, x2 = xb + dx;
        bool ok = ((unsigned)z2 < 48u) & ((unsigned)y2 < 48u) & ((unsigned)x2 < 48u);
        int nvox = (z2 * 48 + y2) * 48 + x2;
        const __bf16* bptr = xin + (size_t)(ok ? nvox : 0) * CP + hi * 16;

        // preload all 4 activation fragments for this tap (one load clause)
        v16bf bfr[4];
#pragma unroll
        for (int ks = 0; ks < 4; ++ks) {
            v16bf b = *(const v16bf*)(bptr + ks * 32);
            if (!ok) {
#pragma unroll
                for (int t = 0; t < 16; ++t) b[t] = (__bf16)0.0f;
            }
            bfr[ks] = b;
        }

        // software-pipelined weight fragments: ds_load(t+1) ahead of wmma(t)
        const __bf16* wlds = &smem[buf * TAP_ELEMS + lane * 16];
        v16bf acur = *(const v16bf*)(wlds);
#pragma unroll
        for (int t = 0; t < 32; ++t) {
            v16bf anext;
            if (t < 31) anext = *(const v16bf*)(wlds + (t + 1) * 512);
            int ks = t >> 3, m = t & 7;
            acc[m] = __builtin_amdgcn_wmma_f32_16x16x32_bf16(
                false, acur, false, bfr[ks], (short)0, acc[m], false, false);
            if (t < 31) acur = anext;
        }

        asm volatile("s_wait_asynccnt 0x0" ::: "memory");
        __syncthreads();   // next-tap weights visible; this buffer safe to reuse
    }

    // C layout: vgpr r, lanes 0-15 -> (co = mt*16+r, vox=lane); lanes 16-31 -> co+8
#pragma unroll
    for (int m = 0; m < 8; ++m) {
        int cob = m * 16 + hi * 8;
        float* dst = y + (size_t)vox * CP + cob;
#pragma unroll
        for (int r = 0; r < 8; ++r) {
            float v = acc[m][r] + b0[cob + r];
            dst[r] = v > 0.0f ? v : 0.0f;
        }
    }
}

// ---------------- per-channel mean / rsqrt(var+eps) ----------------
__global__ void k_stats(const float* __restrict__ y, float* __restrict__ meanv,
                        float* __restrict__ rsigv) {
    int c = blockIdx.x;
    __shared__ float ss[256], sq[256];
    float s = 0.f, q = 0.f;
    for (int v = threadIdx.x; v < NG; v += 256) {
        float t = y[(size_t)v * CP + c];
        s += t; q += t * t;
    }
    ss[threadIdx.x] = s; sq[threadIdx.x] = q;
    __syncthreads();
    for (int o = 128; o > 0; o >>= 1) {
        if (threadIdx.x < (unsigned)o) {
            ss[threadIdx.x] += ss[threadIdx.x + o];
            sq[threadIdx.x] += sq[threadIdx.x + o];
        }
        __syncthreads();
    }
    if (threadIdx.x == 0) {
        float mu = ss[0] / (float)NG;
        float var = sq[0] / (float)NG - mu * mu;
        meanv[c] = mu;
        rsigv[c] = rsqrtf(var + EPSN);
    }
}

// ---------------- normalize -> bf16 fg [vox][ch] ----------------
__global__ void k_norm(const float* __restrict__ y, const float* __restrict__ meanv,
                       const float* __restrict__ rsigv, __bf16* __restrict__ fg) {
    int idx = blockIdx.x * blockDim.x + threadIdx.x;
    if (idx >= NG * CP) return;
    int c = idx & 127;
    fg[idx] = (__bf16)((y[idx] - meanv[c]) * rsigv[c]);
}

// ---------------- trilinear sample keypoints + ||fk||^2 ----------------
__global__ void k_sample(const __bf16* __restrict__ fg, const float* __restrict__ kpts,
                         float* __restrict__ fk, float* __restrict__ knorm) {
    int n = blockIdx.x * blockDim.x + threadIdx.x;
    if (n >= NK) return;
    float ix = (kpts[n * 3 + 0] + 1.f) * 0.5f * 47.f;
    float iy = (kpts[n * 3 + 1] + 1.f) * 0.5f * 47.f;
    float iz = (kpts[n * 3 + 2] + 1.f) * 0.5f * 47.f;
    float fx = floorf(ix), fy = floorf(iy), fz = floorf(iz);
    float wx = ix - fx, wy = iy - fy, wz = iz - fz;
    int x0 = (int)fx; x0 = x0 < 0 ? 0 : (x0 > 47 ? 47 : x0);
    int y0 = (int)fy; y0 = y0 < 0 ? 0 : (y0 > 47 ? 47 : y0);
    int z0 = (int)fz; z0 = z0 < 0 ? 0 : (z0 > 47 ? 47 : z0);
    int x1 = x0 + 1 > 47 ? 47 : x0 + 1;
    int y1 = y0 + 1 > 47 ? 47 : y0 + 1;
    int z1 = z0 + 1 > 47 ? 47 : z0 + 1;
    int o000 = ((z0 * 48 + y0) * 48 + x0) * CP, o001 = ((z0 * 48 + y0) * 48 + x1) * CP;
    int o010 = ((z0 * 48 + y1) * 48 + x0) * CP, o011 = ((z0 * 48 + y1) * 48 + x1) * CP;
    int o100 = ((z1 * 48 + y0) * 48 + x0) * CP, o101 = ((z1 * 48 + y0) * 48 + x1) * CP;
    int o110 = ((z1 * 48 + y1) * 48 + x0) * CP, o111 = ((z1 * 48 + y1) * 48 + x1) * CP;
    float w000 = (1 - wz) * (1 - wy) * (1 - wx), w001 = (1 - wz) * (1 - wy) * wx;
    float w010 = (1 - wz) * wy * (1 - wx),       w011 = (1 - wz) * wy * wx;
    float w100 = wz * (1 - wy) * (1 - wx),       w101 = wz * (1 - wy) * wx;
    float w110 = wz * wy * (1 - wx),             w111 = wz * wy * wx;
    float qn = 0.f;
    for (int c = 0; c < CP; ++c) {
        float v = w000 * (float)fg[o000 + c] + w001 * (float)fg[o001 + c]
                + w010 * (float)fg[o010 + c] + w011 * (float)fg[o011 + c]
                + w100 * (float)fg[o100 + c] + w101 * (float)fg[o101 + c]
                + w110 * (float)fg[o110 + c] + w111 * (float)fg[o111 + c];
        fk[n * CP + c] = v;
        qn += v * v;
    }
    knorm[n] = qn;
}

// ---------------- pack fk into A-fragments (M=kpt, K=ch) ----------------
__global__ void k_pack_fk(const float* __restrict__ fk, __bf16* __restrict__ fkA) {
    int idx = blockIdx.x * blockDim.x + threadIdx.x;
    if (idx >= 32 * 4 * 512) return;
    int i    = idx & 15;
    int lane = (idx >> 4) & 31;
    int frag = idx >> 9;        // mt*4 + ks
    int ks   = frag & 3;
    int mt   = frag >> 2;
    int hi   = lane >> 4;
    int kp   = mt * 16 + (lane & 15);
    int ch   = ks * 32 + a_klocal(i, hi);
    fkA[idx] = (__bf16)fk[kp * CP + ch];
}

// ---------------- cross GEMM + online softmax + disp reduce ----------------
// softmax(-d2) == softmax(2*cross - ||fk||^2): the per-voxel norm cancels.
__global__ void __launch_bounds__(256) k_attn(const __bf16* __restrict__ fg,
                                              const __bf16* __restrict__ fkA,
                                              const float* __restrict__ knorm,
                                              const float* __restrict__ disp,
                                              float* __restrict__ out) {
    __shared__ float sdisp[NK * 3];
    __shared__ float sknorm[NK];
    for (int t = threadIdx.x; t < NK * 3; t += 256) sdisp[t] = disp[t];
    for (int t = threadIdx.x; t < NK; t += 256) sknorm[t] = knorm[t];
    __syncthreads();

    int lane = threadIdx.x & 31;
    int wave = threadIdx.x >> 5;
    int vtile = (blockIdx.x * 8 + wave) * 16;
    int hi = lane >> 4;
    int l  = lane & 15;
    int vox = vtile + l;

    // B fragments: per-voxel features, reused across all 32 keypoint chunks
    v16bf b[4];
#pragma unroll
    for (int ks = 0; ks < 4; ++ks)
        b[ks] = *(const v16bf*)(fg + (size_t)vox * CP + ks * 32 + hi * 16);

    float run_m = -1e30f, run_l = 0.f, adx = 0.f, ady = 0.f, adz = 0.f;

    for (int mt = 0; mt < 32; ++mt) {
        const __bf16* ap = fkA + (size_t)(mt * 4) * 512 + lane * 16;
        // one load clause for all 4 A-fragments, then 4 back-to-back WMMAs
        v16bf a[4];
#pragma unroll
        for (int ks = 0; ks < 4; ++ks)
            a[ks] = *(const v16bf*)(ap + ks * 512);
        v8f acc;
#pragma unroll
        for (int r = 0; r < 8; ++r) acc[r] = 0.f;
#pragma unroll
        for (int ks = 0; ks < 4; ++ks)
            acc = __builtin_amdgcn_wmma_f32_16x16x32_bf16(
                false, a[ks], false, b[ks], (short)0, acc, false, false);

        int mbase = mt * 16 + hi * 8;
        float lg[8];
        float lmax = -1e30f;
#pragma unroll
        for (int r = 0; r < 8; ++r) {
            lg[r] = 2.f * acc[r] - sknorm[mbase + r];
            lmax = fmaxf(lmax, lg[r]);
        }
        float cmax = fmaxf(lmax, __shfl_xor(lmax, 16, 32)); // combine kpt halves
        float nm = fmaxf(run_m, cmax);
        float scale = __expf(run_m - nm);
        float ls = 0.f, dx = 0.f, dy = 0.f, dz = 0.f;
#pragma unroll
        for (int r = 0; r < 8; ++r) {
            float e = __expf(lg[r] - nm);
            int m = mbase + r;
            ls += e;
            dx += e * sdisp[m * 3 + 0];
            dy += e * sdisp[m * 3 + 1];
            dz += e * sdisp[m * 3 + 2];
        }
        ls += __shfl_xor(ls, 16, 32);
        dx += __shfl_xor(dx, 16, 32);
        dy += __shfl_xor(dy, 16, 32);
        dz += __shfl_xor(dz, 16, 32);
        run_l = run_l * scale + ls;
        adx = adx * scale + dx;
        ady = ady * scale + dy;
        adz = adz * scale + dz;
        run_m = nm;
    }
    if (hi == 0) {
        float inv = 1.f / run_l;
        out[0 * NG + vox] = adx * inv;
        out[1 * NG + vox] = ady * inv;
        out[2 * NG + vox] = adz * inv;
    }
}

extern "C" void kernel_launch(void* const* d_in, const int* in_sizes, int n_in,
                              void* d_out, int out_size, void* d_ws, size_t ws_size,
                              hipStream_t stream) {
    const float* kpts  = (const float*)d_in[0];
    const float* disp  = (const float*)d_in[1];
    const float* feats = (const float*)d_in[2];
    const float* W0    = (const float*)d_in[3];
    const float* b0    = (const float*)d_in[4];
    float* out = (float*)d_out;

    char* ws = (char*)d_ws;
    size_t off = 0;
    auto carve = [&](size_t bytes) {
        void* p = ws + off;
        off += (bytes + 255) & ~(size_t)255;
        return p;
    };
    __bf16* xin   = (__bf16*)carve((size_t)NG * CP * 2); // reused as fg after conv
    __bf16* Wp    = (__bf16*)carve((size_t)27 * 4 * 8 * 512 * 2);
    float*  y     = (float*) carve((size_t)NG * CP * 4);
    float*  fk    = (float*) carve((size_t)NK * CP * 4);
    __bf16* fkA   = (__bf16*)carve((size_t)32 * 4 * 512 * 2);
    float*  knorm = (float*) carve((size_t)NK * 4);
    float*  meanv = (float*) carve((size_t)CP * 4);
    float*  rsigv = (float*) carve((size_t)CP * 4);
    (void)ws_size; (void)in_sizes; (void)n_in; (void)out_size;

    k_pack_w  <<<(27 * 4 * 8 * 512 + 255) / 256, 256, 0, stream>>>(W0, Wp);
    k_build_in<<<(NG * CP + 255) / 256, 256, 0, stream>>>(feats, xin);
    k_conv    <<<NG / (16 * 8), 256, 0, stream>>>(xin, Wp, b0, y);
    k_stats   <<<CP, 256, 0, stream>>>(y, meanv, rsigv);
    k_norm    <<<(NG * CP + 255) / 256, 256, 0, stream>>>(y, meanv, rsigv, xin);
    k_sample  <<<(NK + 255) / 256, 256, 0, stream>>>(xin, kpts, fk, knorm);
    k_pack_fk <<<(32 * 4 * 512 + 255) / 256, 256, 0, stream>>>(fk, fkA);
    k_attn    <<<NG / (16 * 8), 256, 0, stream>>>(xin, fkA, knorm, disp, out);
}
